// CausalMultiHeadSelfAttention_32169305047422
// MI455X (gfx1250) — matrix operand order repaired and checked by reading, not compiled, not run on hardware
//
#include <hip/hip_runtime.h>
#include <hip/hip_bf16.h>

// ---------------------------------------------------------------------------
// Causal MHA for MI455X (gfx1250, wave32).  All matmuls via
// v_wmma_f32_16x16x32_bf16 (f32 accumulate).  Compute-bound overall, but the
// flash-attention K/V stream and the GEMM B-matrix are shared across all 8
// waves of a workgroup through LDS, staged with the CDNA5 async-to-LDS path
// (ASYNCcnt + s_wait_asynccnt), double-buffered against the WMMA pipeline.
// ---------------------------------------------------------------------------

typedef __bf16 bf16_t;
typedef __attribute__((ext_vector_type(16))) __bf16 v16bf;
typedef __attribute__((ext_vector_type(8)))  __bf16 v8bf;
typedef __attribute__((ext_vector_type(8)))  float  v8f;
typedef __attribute__((ext_vector_type(4)))  int    v4i_t;

#define B_  4
#define S_  2048
#define D_  1024
#define H_  16
#define DH_ 64

#define AS1 __attribute__((address_space(1)))
#define AS3 __attribute__((address_space(3)))

#if __has_builtin(__builtin_amdgcn_global_load_async_to_lds_b128)
#define HAS_ASYNC_LDS 1
#else
#define HAS_ASYNC_LDS 0
#endif

// 16-byte global->LDS copy: async (ASYNCcnt) when available, sync fallback.
__device__ __forceinline__ void async_copy16(bf16_t* lds, const bf16_t* g) {
#if HAS_ASYNC_LDS
  __builtin_amdgcn_global_load_async_to_lds_b128((AS1 v4i_t*)g, (AS3 v4i_t*)lds,
                                                 0, 0);
#else
  *(uint4*)lds = *(const uint4*)g;
#endif
}

template <int N>
__device__ __forceinline__ void wait_asynccnt() {
#if HAS_ASYNC_LDS
#if __has_builtin(__builtin_amdgcn_s_wait_asynccnt)
  __builtin_amdgcn_s_wait_asynccnt((short)N);
#else
  asm volatile("s_wait_asynccnt %0" ::"i"(N) : "memory");
#endif
#endif
}

union ABFrag { v16bf v; v8bf h[2]; };

// A-matrix fragment (16x32 bf16): lane holds row M=lane&15; elements
// j -> k = half*16 + j.
__device__ __forceinline__ v16bf load_a_frag(const bf16_t* rowptr, int koff, int half) {
  ABFrag f;
  f.v = *(const v16bf*)(rowptr + koff + half * 16);
  return f.v;
}

__device__ __forceinline__ v8f wmma_bf16(v16bf a, v16bf b, v8f c) {
  return __builtin_amdgcn_wmma_f32_16x16x32_bf16(false, a, false, b, (short)0, c,
                                                 false, false);
}

// ---------------------------------------------------------------------------
// f32 -> bf16 conversion
// ---------------------------------------------------------------------------
__global__ void cvt_f32_bf16_kernel(const float* __restrict__ in,
                                    bf16_t* __restrict__ out, int n) {
  int i = blockIdx.x * blockDim.x + threadIdx.x;
  if (i < n) out[i] = (bf16_t)in[i];
}

// ---------------------------------------------------------------------------
// RoPE cos/sin tables: [S][32]
// ---------------------------------------------------------------------------
__global__ void rope_table_kernel(float* __restrict__ ct, float* __restrict__ st) {
  int i = blockIdx.x * blockDim.x + threadIdx.x;
  if (i < S_ * 32) {
    int s = i >> 5;
    int f = i & 31;
    float freq = __expf(-(float)(2 * f) * (9.210340371976184f / 64.0f));  // theta^(-2f/DH)
    float ang = (float)s * freq;
    ct[i] = cosf(ang);
    st[i] = sinf(ang);
  }
}

// ---------------------------------------------------------------------------
// Cooperative B-tile stage: 128 cols x 32 k (8KB) into LDS, layout [n][32k]
// so each WMMA B-fragment is a 32B contiguous per-lane LDS load.
// 256 threads x 2 x 16B async copies.
// ---------------------------------------------------------------------------
__device__ __forceinline__ void stage_b(bf16_t* sb, const bf16_t* W, int n0,
                                        int k0, int tid) {
#pragma unroll
  for (int u = 0; u < 2; ++u) {
    const int chunk = tid + u * 256;   // 0..511
    const int n = chunk >> 2;          // 0..127
    const int c = chunk & 3;           // 16B chunk within the 32-k row
    async_copy16(sb + n * 32 + c * 8, W + (size_t)(n0 + n) * D_ + k0 + c * 8);
  }
}

// ---------------------------------------------------------------------------
// QKV GEMM (bf16 WMMA) + fused RoPE epilogue.
// Block = 8 waves = 128 rows x 128 cols; wave w: rows m0+16w, 8 N-tiles.
// B tile staged in LDS (async, double-buffered), shared by all waves.
// Q,K get RoPE -> [B,H,S,DH]; V -> transposed [B,H,DH,S].
// ---------------------------------------------------------------------------
__global__ __launch_bounds__(256) void qkv_rope_kernel(
    const bf16_t* __restrict__ xbf, const bf16_t* __restrict__ wqkv,
    const float* __restrict__ ctab, const float* __restrict__ stab,
    bf16_t* __restrict__ qb, bf16_t* __restrict__ kb, bf16_t* __restrict__ vtb) {
  __shared__ __align__(16) bf16_t sB[2][128 * 32];
  const int tid  = threadIdx.x;
  const int lane = tid & 31;
  const int w    = tid >> 5;
  const int half = lane >> 4;
  const int nl   = lane & 15;
  const int m0 = blockIdx.y * 128 + w * 16;
  const int n0 = blockIdx.x * 128;

  const bf16_t* arow = xbf + (size_t)(m0 + nl) * D_;
  v8f acc[8];
  v8f z = {};
#pragma unroll
  for (int t = 0; t < 8; ++t) acc[t] = z;

  stage_b(sB[0], wqkv, n0, 0, tid);
  for (int it = 0; it < 32; ++it) {
    const int k0 = it * 32;
    if (it + 1 < 32) {
      stage_b(sB[(it + 1) & 1], wqkv, n0, k0 + 32, tid);
      wait_asynccnt<2>();   // previous stage (2 ops) complete; new one in flight
    } else {
      wait_asynccnt<0>();
    }
    __syncthreads();
    const bf16_t* sb = sB[it & 1];
    v16bf a = load_a_frag(arow, k0, half);
#pragma unroll
    for (int t = 0; t < 8; ++t) {
      v16bf b = *(const v16bf*)(sb + (t * 16 + nl) * 32 + half * 16);
      acc[t] = wmma_bf16(a, b, acc[t]);
    }
    __syncthreads();
  }

  // Epilogue: D element r <-> (M = r + half*8, N = nl)
#pragma unroll
  for (int t = 0; t < 8; ++t) {
    const int e0     = n0 + t * 16;
    const int region = e0 >> 10;              // 0=Q 1=K 2=V (uniform per tile)
    const int eh     = (e0 & 1023) + nl;
    const int h      = eh >> 6;
    const int dh     = eh & 63;
    const int i      = dh >> 1;
#pragma unroll
    for (int r = 0; r < 8; ++r) {
      const int row = m0 + half * 8 + r;
      const int b   = row >> 11;
      const int s   = row & (S_ - 1);
      float val = acc[t][r];
      if (region < 2) {  // RoPE
        float cs = ctab[s * 32 + i];
        float sn = stab[s * 32 + i];
        float p  = __shfl_xor(val, 1, 32);
        val = (dh & 1) ? fmaf(p, sn, val * cs)
                       : fmaf(p, -sn, val * cs);
      }
      if (region == 2) {
        vtb[(((size_t)b * H_ + h) * DH_ + dh) * S_ + s] = (bf16_t)val;
      } else {
        bf16_t* dst = (region == 0) ? qb : kb;
        dst[(((size_t)b * H_ + h) * S_ + s) * DH_ + dh] = (bf16_t)val;
      }
    }
  }
}

// ---------------------------------------------------------------------------
// Flash attention.  Block = 8 waves = 128 q rows; block-uniform k-loop so the
// 32x64 K tile and 64x32 V^T tile are staged ONCE per block into LDS via the
// async path (double-buffered), then consumed by all waves.  Score->P operand
// transpose goes through a per-wave 16x32 LDS tile (s_wait_dscnt fence).
// ---------------------------------------------------------------------------
__global__ __launch_bounds__(256) void flash_attn_kernel(
    const bf16_t* __restrict__ qb, const bf16_t* __restrict__ kb,
    const bf16_t* __restrict__ vtb, bf16_t* __restrict__ ob) {
  __shared__ __align__(16) bf16_t sK[2][32 * 64];  // [kpos][dh]
  __shared__ __align__(16) bf16_t sV[2][64 * 32];  // [dh][kpos]
  __shared__ __align__(16) bf16_t sP[8][16 * 32];  // per-wave P tile
  const int tid  = threadIdx.x;
  const int lane = tid & 31;
  const int w    = tid >> 5;
  const int half = lane >> 4;
  const int nl   = lane & 15;
  const int b   = blockIdx.z;
  const int h   = blockIdx.y;
  const int q0b = blockIdx.x * 128;
  const int q0  = q0b + w * 16;

  const bf16_t* Qh = qb  + ((size_t)b * H_ + h) * S_ * DH_;
  const bf16_t* Kh = kb  + ((size_t)b * H_ + h) * S_ * DH_;
  const bf16_t* Vh = vtb + ((size_t)b * H_ + h) * DH_ * S_;
  bf16_t* pw = sP[w];

  // Loop-invariant Q fragments: A(16x64) as two 16x32 frags.
  const bf16_t* qrow = Qh + (size_t)(q0 + nl) * DH_;
  v16bf aq0 = load_a_frag(qrow, 0, half);
  v16bf aq1 = load_a_frag(qrow, 32, half);

  float row_max[8], row_sum[8];
  v8f oacc[4];
  v8f z = {};
#pragma unroll
  for (int r = 0; r < 8; ++r) { row_max[r] = -3.0e30f; row_sum[r] = 0.0f; }
#pragma unroll
  for (int t = 0; t < 4; ++t) oacc[t] = z;

  const int vdh = tid >> 2, vch = tid & 3;  // V staging coords
  const int nsteps = (q0b >> 5) + 4;        // keys [0, q0b+128), 32 per step

  // prologue stage: K tile is a contiguous 4KB block; V tile gathers 64 rows.
  async_copy16(sK[0] + tid * 8, Kh + tid * 8);
  async_copy16(sV[0] + vdh * 32 + vch * 8, Vh + (size_t)vdh * S_ + vch * 8);

  for (int step = 0; step < nsteps; ++step) {
    const int k0 = step * 32;
    if (step + 1 < nsteps) {
      const int kn = k0 + 32;
      bf16_t* skn = sK[(step + 1) & 1];
      bf16_t* svn = sV[(step + 1) & 1];
      async_copy16(skn + tid * 8, Kh + (size_t)kn * DH_ + tid * 8);
      async_copy16(svn + vdh * 32 + vch * 8, Vh + (size_t)vdh * S_ + kn + vch * 8);
      wait_asynccnt<2>();
    } else {
      wait_asynccnt<0>();
    }
    __syncthreads();
    const bf16_t* sk = sK[step & 1];
    const bf16_t* sv = sV[step & 1];

    // ---- scores: two 16x16 tiles, 64-deep contraction, B from LDS ----
    v8f sc[2];
#pragma unroll
    for (int t2 = 0; t2 < 2; ++t2) {
      const bf16_t* kr = sk + (t2 * 16 + nl) * 64;
      v16bf bk0 = *(const v16bf*)(kr + half * 16);       // dh 0..31
      v16bf bk1 = *(const v16bf*)(kr + 32 + half * 16);  // dh 32..63
      v8f c = z;
      c = wmma_bf16(aq0, bk0, c);
      c = wmma_bf16(aq1, bk1, c);
      sc[t2] = c;
    }

    // ---- scale + causal mask ----
#pragma unroll
    for (int t2 = 0; t2 < 2; ++t2) {
      const int kp = k0 + t2 * 16 + nl;
#pragma unroll
      for (int r = 0; r < 8; ++r) {
        const int qp = q0 + half * 8 + r;
        float v = sc[t2][r] * 0.125f;  // 1/sqrt(64)
        sc[t2][r] = (kp <= qp) ? v : -3.0e30f;
      }
    }

    // ---- streaming softmax (row stats replicated per 16-lane group) ----
#pragma unroll
    for (int r = 0; r < 8; ++r) {
      float mx = fmaxf(sc[0][r], sc[1][r]);
#pragma unroll
      for (int off = 1; off < 16; off <<= 1) mx = fmaxf(mx, __shfl_xor(mx, off, 32));
      const float nm   = fmaxf(row_max[r], mx);
      const float corr = __expf(row_max[r] - nm);
      const float p0   = __expf(sc[0][r] - nm);
      const float p1   = __expf(sc[1][r] - nm);
      float s2 = p0 + p1;
#pragma unroll
      for (int off = 1; off < 16; off <<= 1) s2 += __shfl_xor(s2, off, 32);
      row_sum[r] = row_sum[r] * corr + s2;
      row_max[r] = nm;
#pragma unroll
      for (int t = 0; t < 4; ++t) oacc[t][r] *= corr;
      const int m = half * 8 + r;
      pw[m * 32 + nl]      = (bf16_t)p0;
      pw[m * 32 + 16 + nl] = (bf16_t)p1;
    }

    // DS ops are in-order per wave; fence keeps the compiler honest too.
    asm volatile("s_wait_dscnt 0" ::: "memory");

    // ---- P.V: A = P (16x32, LDS), B = V^T rows (32B contiguous, LDS) ----
    v16bf ap = load_a_frag(pw + (size_t)nl * 32, 0, half);
#pragma unroll
    for (int t = 0; t < 4; ++t) {
      v16bf bv = *(const v16bf*)(sv + (t * 16 + nl) * 32 + half * 16);
      oacc[t] = wmma_bf16(ap, bv, oacc[t]);
    }
    __syncthreads();
  }

  // ---- normalize + write O (bf16, [B,S,D]) for the final projection ----
#pragma unroll
  for (int t = 0; t < 4; ++t) {
#pragma unroll
    for (int r = 0; r < 8; ++r) {
      const int qp = q0 + half * 8 + r;
      float val = oacc[t][r] / row_sum[r];
      ob[((size_t)b * S_ + qp) * D_ + h * DH_ + t * 16 + nl] = (bf16_t)val;
    }
  }
}

// ---------------------------------------------------------------------------
// Output projection: d_out[row, e] = sum_d O[row,d] * Wo[e,d]   (f32 out)
// Same staged-B structure as the QKV GEMM.
// ---------------------------------------------------------------------------
__global__ __launch_bounds__(256) void oproj_kernel(
    const bf16_t* __restrict__ obf, const bf16_t* __restrict__ wobf,
    float* __restrict__ out) {
  __shared__ __align__(16) bf16_t sB[2][128 * 32];
  const int tid  = threadIdx.x;
  const int lane = tid & 31;
  const int w    = tid >> 5;
  const int half = lane >> 4;
  const int nl   = lane & 15;
  const int m0 = blockIdx.y * 128 + w * 16;
  const int n0 = blockIdx.x * 128;

  const bf16_t* arow = obf + (size_t)(m0 + nl) * D_;
  v8f acc[8];
  v8f z = {};
#pragma unroll
  for (int t = 0; t < 8; ++t) acc[t] = z;

  stage_b(sB[0], wobf, n0, 0, tid);
  for (int it = 0; it < 32; ++it) {
    const int k0 = it * 32;
    if (it + 1 < 32) {
      stage_b(sB[(it + 1) & 1], wobf, n0, k0 + 32, tid);
      wait_asynccnt<2>();
    } else {
      wait_asynccnt<0>();
    }
    __syncthreads();
    const bf16_t* sb = sB[it & 1];
    v16bf a = load_a_frag(arow, k0, half);
#pragma unroll
    for (int t = 0; t < 8; ++t) {
      v16bf b = *(const v16bf*)(sb + (t * 16 + nl) * 32 + half * 16);
      acc[t] = wmma_bf16(a, b, acc[t]);
    }
    __syncthreads();
  }

#pragma unroll
  for (int t = 0; t < 8; ++t) {
#pragma unroll
    for (int r = 0; r < 8; ++r) {
      const int row = m0 + half * 8 + r;
      out[(size_t)row * D_ + n0 + t * 16 + nl] = acc[t][r];
    }
  }
}

// ---------------------------------------------------------------------------
// Workspace layout (bytes):
//   [ 0,16M)  x_bf16          [16M,22M) wqkv_bf16       [22M,24M) wo_bf16
//   [24M,40M) Q [B,H,S,DH]    [40M,56M) K [B,H,S,DH]    [56M,72M) Vt [B,H,DH,S]
//   [72M,88M) O bf16 [B*S,D]  [88M,..)  cos table       [89M,..)  sin table
// ---------------------------------------------------------------------------
extern "C" void kernel_launch(void* const* d_in, const int* in_sizes, int n_in,
                              void* d_out, int out_size, void* d_ws, size_t ws_size,
                              hipStream_t stream) {
  (void)in_sizes; (void)n_in; (void)out_size; (void)ws_size;
  const float* x  = (const float*)d_in[0];
  const float* wq = (const float*)d_in[1];
  const float* wk = (const float*)d_in[2];
  const float* wv = (const float*)d_in[3];
  const float* wo = (const float*)d_in[4];
  float* out = (float*)d_out;

  char* ws = (char*)d_ws;
  const size_t MB = 1024u * 1024u;
  bf16_t* xbf  = (bf16_t*)(ws + 0 * MB);
  bf16_t* wqkv = (bf16_t*)(ws + 16 * MB);
  bf16_t* wobf = (bf16_t*)(ws + 22 * MB);
  bf16_t* qb   = (bf16_t*)(ws + 24 * MB);
  bf16_t* kb   = (bf16_t*)(ws + 40 * MB);
  bf16_t* vtb  = (bf16_t*)(ws + 56 * MB);
  bf16_t* ob   = (bf16_t*)(ws + 72 * MB);
  float*  ctab = (float*) (ws + 88 * MB);
  float*  stab = (float*) (ws + 89 * MB);

  const int NX = B_ * S_ * D_;   // 8388608
  const int NW = D_ * D_;        // 1048576

  cvt_f32_bf16_kernel<<<(NX + 255) / 256, 256, 0, stream>>>(x, xbf, NX);
  cvt_f32_bf16_kernel<<<(NW + 255) / 256, 256, 0, stream>>>(wq, wqkv, NW);
  cvt_f32_bf16_kernel<<<(NW + 255) / 256, 256, 0, stream>>>(wk, wqkv + NW, NW);
  cvt_f32_bf16_kernel<<<(NW + 255) / 256, 256, 0, stream>>>(wv, wqkv + 2 * NW, NW);
  cvt_f32_bf16_kernel<<<(NW + 255) / 256, 256, 0, stream>>>(wo, wobf, NW);

  rope_table_kernel<<<(S_ * 32 + 255) / 256, 256, 0, stream>>>(ctab, stab);

  // QKV + RoPE: M = 8192 (64 x 128), N = 3072 (24 x 128)
  qkv_rope_kernel<<<dim3(24, 64), 256, 0, stream>>>(xbf, wqkv, ctab, stab,
                                                    qb, kb, vtb);

  // Flash attention: (S/128, H, B) blocks of 8 waves
  flash_attn_kernel<<<dim3(S_ / 128, H_, B_), 256, 0, stream>>>(qb, kb, vtb, ob);

  // Output projection: M = 8192 (64 x 128), N = 1024 (8 x 128)
  oproj_kernel<<<dim3(8, 64), 256, 0, stream>>>(ob, wobf, out);
}
